// Model_31885837205644
// MI455X (gfx1250) — matrix operand order repaired and checked by reading, not compiled, not run on hardware
//
#include <hip/hip_runtime.h>
#include <hip/hip_bf16.h>
#include <stdint.h>

// ---------------------------------------------------------------------------
// CDNA5 (gfx1250) sparse 3D U-Net forward.
// bf16 activations; core GEMMs: v_wmma_f32_16x16x32_bf16 (wave32).
// Subm conv: 32-row M tile, LDS double-buffered K chunks (single buffer when
// one chunk), depth-2 register pipeline on fragments, dual accumulators
// sharing each B fragment, vectorized (b128) LDS gather.
// ---------------------------------------------------------------------------

typedef __attribute__((ext_vector_type(16))) __bf16 v16bf;
typedef __attribute__((ext_vector_type(8)))  float  v8f;

#define KCH   512        // K-chunk staged in LDS (power of 2)
#define EPSBN 1e-3f

__device__ __forceinline__ unsigned short f2bf(float f) {
  unsigned int u = __float_as_uint(f);
  u += 0x7fffu + ((u >> 16) & 1u);   // round-to-nearest-even
  return (unsigned short)(u >> 16);
}
__device__ __forceinline__ float bf2f(unsigned short h) {
  return __uint_as_float(((unsigned int)h) << 16);
}

// A-fragment (16x32 bf16): lane holds row lane&15;
// elems 0..15 <-> K = half*16+0..15 (contiguous, matches B)
__device__ __forceinline__ v16bf load_a_frag(const unsigned short* lo,
                                             const unsigned short* hi) {
  union { uint4 u[2]; v16bf v; } t;
  t.u[0] = *(const uint4*)lo;
  t.u[1] = *(const uint4*)(lo + 8); (void)hi;
  return t.v;
}

// ---------------------------------------------------------------------------
// Submanifold conv + fused BN/ReLU, bf16 in / bf16 out.
// wt bf16 transposed: wt[d][K], K = k*CIN + c, zero-padded to Kpad.
// ---------------------------------------------------------------------------
template <int CIN>
__global__ __launch_bounds__(256)
void subm_wmma_kernel(const unsigned short* __restrict__ x,
                      const int* __restrict__ nbr,
                      const unsigned short* __restrict__ wt,
                      const float* __restrict__ bg, const float* __restrict__ bb,
                      const float* __restrict__ bm, const float* __restrict__ bv,
                      unsigned short* __restrict__ out, int N, int Cout, int Kpad)
{
  constexpr int KTOT = 27 * CIN;
  extern __shared__ char smem[];
  const int nchunks = (Kpad + KCH - 1) / KCH;
  unsigned short* A0  = (unsigned short*)smem;                       // [32][KCH]
  unsigned short* A1  = (nchunks > 1) ? (A0 + 32 * KCH) : A0;        // [32][KCH]
  int*            snb = (int*)(A0 + (nchunks > 1 ? 2 : 1) * 32 * KCH);
  const int n0  = blockIdx.x << 5;
  const int tid = threadIdx.x, nth = blockDim.x;

  for (int i = tid; i < 27 * 32; i += nth) {
    int k = i >> 5, r = i & 31, n = n0 + r;
    snb[i] = (n < N) ? nbr[k * N + n] : -1;
  }

  const int lane = tid & 31, half = lane >> 4, lr = lane & 15;
  const int col  = ((tid >> 5) << 4) + lr;

  auto gather = [&](int t, unsigned short* buf) {
    const int kb = t * KCH;
    const int kchunk = (Kpad - kb < KCH) ? (Kpad - kb) : KCH;
    if constexpr (CIN % 16 == 0) {
      // 8 bf16 per thread: one b128 global load + one b128 LDS store
      const int vch = kchunk >> 3;
      for (int e = tid; e < 32 * (KCH >> 3); e += nth) {
        int kv = e & ((KCH >> 3) - 1);
        if (kv >= vch) continue;
        int r = e >> 6;                       // KCH/8 == 64
        int K = kb + (kv << 3);
        uint4 val; val.x = val.y = val.z = val.w = 0u;
        if (K < KTOT) {
          int k = K / CIN, c = K - k * CIN;   // compile-time divisor
          int nn = snb[(k << 5) + r];
          if (nn >= 0) val = *(const uint4*)(x + (size_t)nn * CIN + c);
        }
        *(uint4*)(buf + (r << 9) + (kv << 3)) = val;
      }
    } else {
      for (int e = tid; e < 32 * KCH; e += nth) {
        int kk = e & (KCH - 1);
        if (kk >= kchunk) continue;
        int r = e >> 9;
        int K = kb + kk;
        unsigned short hv = 0;
        if (K < KTOT) {
          int k = K / CIN, c = K - k * CIN;
          int nn = snb[(k << 5) + r];
          if (nn >= 0) hv = x[(size_t)nn * CIN + c];
        }
        buf[(r << 9) + kk] = hv;
      }
    }
  };

  v8f acc0 = {}, acc1 = {};
  gather(0, A0);
  __syncthreads();

  for (int t = 0; t < nchunks; ++t) {
    unsigned short* bufc = (t & 1) ? A1 : A0;
    unsigned short* bufn = (t & 1) ? A0 : A1;
    if (t + 1 < nchunks) gather(t + 1, bufn);   // overlaps with WMMA below

    const int kb = t * KCH;
    const int kchunk = (Kpad - kb < KCH) ? (Kpad - kb) : KCH;
    const int steps = kchunk >> 5;
    const unsigned short* wrow = wt + (size_t)col * Kpad + kb + half * 16;
    const unsigned short* ar0  = bufc + (lr << 9) + half * 16;
    const unsigned short* ar1  = bufc + ((lr + 16) << 9) + half * 16;

    // depth-2 software pipeline: consumed set was loaded 2 iterations ago
    v16bf b0  = *(const v16bf*)(wrow);
    v16bf a00 = load_a_frag(ar0, ar0 + 16);
    v16bf a10 = load_a_frag(ar1, ar1 + 16);
    const int k1 = (steps > 1) ? 32 : 0;
    v16bf b1  = *(const v16bf*)(wrow + k1);
    v16bf a01 = load_a_frag(ar0 + k1, ar0 + k1 + 16);
    v16bf a11 = load_a_frag(ar1 + k1, ar1 + k1 + 16);
    for (int s = 0; s < steps; ++s) {
      const int kp = (s + 2 < steps) ? ((s + 2) << 5) : (s << 5);
      __builtin_prefetch(wrow + kp + 512, 0, 3);      // warm B stream (L2->near)
      v16bf bp  = *(const v16bf*)(wrow + kp);
      v16bf ap0 = load_a_frag(ar0 + kp, ar0 + kp + 16);
      v16bf ap1 = load_a_frag(ar1 + kp, ar1 + kp + 16);
      acc0 = __builtin_amdgcn_wmma_f32_16x16x32_bf16(
                 false, a00, false, b0, (short)0, acc0, false, false);
      acc1 = __builtin_amdgcn_wmma_f32_16x16x32_bf16(
                 false, a10, false, b0, (short)0, acc1, false, false);
      b0 = b1;  a00 = a01; a10 = a11;
      b1 = bp;  a01 = ap0; a11 = ap1;
    }
    __syncthreads();
  }

  // Epilogue BN+ReLU -> bf16. C/D layout: lane -> col, VGPR i -> row i+half*8.
  float s  = bg[col] * rsqrtf(bv[col] + EPSBN);
  float mm = bm[col], bo = bb[col];
#pragma unroll
  for (int i = 0; i < 8; ++i) {
    int r = (half << 3) + i;
    int n = n0 + r;
    if (n < N)
      out[(size_t)n * Cout + col] = f2bf(fmaxf((acc0[i] - mm) * s + bo, 0.f));
    int n2 = n0 + 16 + r;
    if (n2 < N)
      out[(size_t)n2 * Cout + col] = f2bf(fmaxf((acc1[i] - mm) * s + bo, 0.f));
  }
}

// ---------------------------------------------------------------------------
// Rulebook scatter conv: per offset k, out[sidx[k][m]] += x[gidx[k][m]] @ w[k]
// x bf16; out fp32 accumulator (compact [n_out][Cout], atomic adds).
// wt2 bf16 layout [27][Cout][CPAD]. B fragments preloaded before gather sync.
// ---------------------------------------------------------------------------
template <int CIN>
__global__ __launch_bounds__(256)
void scatter_wmma_kernel(const unsigned short* __restrict__ x,
                         const int* __restrict__ gidx, const int* __restrict__ sidx,
                         const unsigned short* __restrict__ wt2,
                         float* __restrict__ out, int Mx, int Cout, int tiles)
{
  constexpr int CPAD = (CIN + 31) & ~31;
  constexpr int NB   = CPAD / 32;
  __shared__ unsigned short Ah[16 * CPAD];
  __shared__ int sg[16], ss[16];
  const int k  = blockIdx.x / tiles;
  const int m0 = (blockIdx.x - k * tiles) << 4;
  const int tid = threadIdx.x, nth = blockDim.x;

  for (int r = tid; r < 16; r += nth) {
    int m = m0 + r;
    sg[r] = (m < Mx) ? gidx[k * Mx + m] : -1;
    ss[r] = (m < Mx) ? sidx[k * Mx + m] : -1;
  }

  const int lane = tid & 31, half = lane >> 4, lr = lane & 15;
  const int col  = ((tid >> 5) << 4) + lr;

  // preload all B fragments (independent of LDS) to hide weight latency
  const unsigned short* wrow = wt2 + ((size_t)(k * Cout + col)) * CPAD + half * 16;
  v16bf bfr[NB];
#pragma unroll
  for (int s = 0; s < NB; ++s) bfr[s] = *(const v16bf*)(wrow + (s << 5));

  __syncthreads();
  for (int e = tid; e < 16 * (CPAD >> 3); e += nth) {
    int r = e / (CPAD >> 3), kv = e - r * (CPAD >> 3);   // compile-time divisor
    int c = kv << 3;
    int nn = sg[r];
    uint4 val; val.x = val.y = val.z = val.w = 0u;
    if (nn >= 0 && c < CIN) val = *(const uint4*)(x + (size_t)nn * CIN + c);
    *(uint4*)(Ah + r * CPAD + c) = val;
  }
  __syncthreads();

  v8f acc = {};
  const unsigned short* arow = Ah + lr * CPAD + half * 16;
#pragma unroll
  for (int s = 0; s < NB; ++s) {
    v16bf a = load_a_frag(arow + (s << 5), arow + (s << 5) + 16);
    acc = __builtin_amdgcn_wmma_f32_16x16x32_bf16(
              false, a, false, bfr[s], (short)0, acc, false, false);
  }
#pragma unroll
  for (int i = 0; i < 8; ++i) {
    int so = ss[(half << 3) + i];
    if (so >= 0) atomicAdd(&out[(size_t)so * Cout + col], acc[i]);
  }
}

// ---------------------------------------------------------------------------
// Helper kernels
// ---------------------------------------------------------------------------
__global__ void zero_kernel(float4* p, long long n4) {
  long long i = (long long)blockIdx.x * blockDim.x + threadIdx.x;
  if (i < n4) { float4 z; z.x = z.y = z.z = z.w = 0.f; p[i] = z; }
}

// fp32 accumulator -> BN+ReLU -> bf16 (strided dst for concat right half)
__global__ void bnrelu_store_kernel(const float* __restrict__ acc,
                                    const float* __restrict__ g, const float* __restrict__ b,
                                    const float* __restrict__ m, const float* __restrict__ v,
                                    unsigned short* __restrict__ dst,
                                    int N, int C, int ld, int coloff) {
  long long i = (long long)blockIdx.x * blockDim.x + threadIdx.x;
  if (i >= (long long)N * C) return;
  int n = (int)(i / C), c = (int)(i - (long long)n * C);
  float s = g[c] * rsqrtf(v[c] + EPSBN);
  float val = (acc[i] - m[c]) * s + b[c];
  dst[(size_t)n * ld + coloff + c] = f2bf(fmaxf(val, 0.f));
}

// bf16 column-block copy, 8 elems (uint4) per thread; Cv = C/8, ldv = ld/8
__global__ void copy_cols_kernel(const uint4* __restrict__ src, uint4* __restrict__ dst,
                                 long long nvec, int Cv, int ldv) {
  long long i = (long long)blockIdx.x * blockDim.x + threadIdx.x;
  if (i >= nvec) return;
  int n = (int)(i / Cv), c = (int)(i - (long long)n * Cv);
  dst[(size_t)n * ldv + c] = src[i];
}

__global__ void f2bf_kernel(const float* __restrict__ src, unsigned short* __restrict__ dst,
                            long long n) {
  long long i = (long long)blockIdx.x * blockDim.x + threadIdx.x;
  if (i < n) dst[i] = f2bf(src[i]);
}

__global__ void linear_kernel(const unsigned short* __restrict__ x,
                              const float* __restrict__ w, const float* __restrict__ b,
                              float* __restrict__ out, int N) {
  long long i = (long long)blockIdx.x * blockDim.x + threadIdx.x;
  if (i >= (long long)N * 20) return;
  int n = (int)(i / 20), j = (int)(i - (long long)n * 20);
  const unsigned short* xr = x + (size_t)n * 16;
  const float* wr = w + j * 16;
  float s = b[j];
#pragma unroll
  for (int c = 0; c < 16; ++c) s += bf2f(xr[c]) * wr[c];
  out[i] = s;
}

// w[27][Cin][Cout] fp32 -> wt[Cout][Kpad] bf16 (K = k*Cin + c, zero padded)
__global__ void wt_full_kernel(const float* __restrict__ w, unsigned short* __restrict__ wt,
                               int Cin, int Cout, int Kpad) {
  long long i = (long long)blockIdx.x * blockDim.x + threadIdx.x;
  if (i >= (long long)Cout * Kpad) return;
  int d = (int)(i / Kpad), K = (int)(i - (long long)d * Kpad);
  float f = (K < 27 * Cin) ? w[(size_t)K * Cout + d] : 0.f;
  wt[i] = f2bf(f);
}

// w[27][Cin][Cout] fp32 -> wt2[27][Cout][Cpad] bf16 (zero padded in c)
__global__ void wt_perk_kernel(const float* __restrict__ w, unsigned short* __restrict__ wt,
                               int Cin, int Cout, int Cpad) {
  long long i = (long long)blockIdx.x * blockDim.x + threadIdx.x;
  if (i >= 27LL * Cout * Cpad) return;
  int k = (int)(i / ((long long)Cout * Cpad));
  long long rem = i - (long long)k * Cout * Cpad;
  int d = (int)(rem / Cpad), c = (int)(rem - (long long)d * Cpad);
  float f = (c < Cin) ? w[((size_t)k * Cin + c) * Cout + d] : 0.f;
  wt[i] = f2bf(f);
}

// ---------------------------------------------------------------------------
// Host orchestration
// ---------------------------------------------------------------------------
extern "C" void kernel_launch(void* const* d_in, const int* in_sizes, int n_in,
                              void* d_out, int out_size, void* d_ws, size_t ws_size,
                              hipStream_t stream) {
  (void)n_in; (void)out_size; (void)ws_size;
  const int P[7] = {16, 32, 48, 64, 80, 96, 112};

  // Input index map — flat in setup_inputs() dict insertion order.
  int idx = 0;
  const int FEAT = idx++;
  const int IW  = idx; idx += 5;
  const int PRE = idx; idx += 60;
  const int DWN = idx; idx += 30;
  const int UPW = idx; idx += 30;
  const int PST = idx; idx += 60;
  const int INR = idx; idx += 10;
  const int LW  = idx++;
  const int LB  = idx++;
  const int NBR = idx; idx += 7;
  const int PIO = idx; idx += 12;

  auto Fp = [&](int i) { return (const float*)d_in[i]; };
  auto Ip = [&](int i) { return (const int*)d_in[i]; };
  auto a32 = [](int x) { return (x + 31) & ~31; };

  int N[7], Mx[6];
  for (int l = 0; l < 7; ++l) N[l] = in_sizes[NBR + l] / 27;
  for (int l = 0; l < 6; ++l) Mx[l] = in_sizes[PIO + 2 * l] / 27;
  const int N0 = N[0];

  // Workspace carve
  char* ws = (char*)d_ws; size_t off = 0;
  auto alloc = [&](size_t bytes) -> void* {
    void* p = ws + off; off = (off + bytes + 255) & ~(size_t)255; return p;
  };
  unsigned short* wt_init = (unsigned short*)alloc((size_t)16 * a32(27 * 3) * 2);
  unsigned short *wt_pre[6][2], *wt_post[6][2], *wt_inner[2], *wt_down[6], *wt_up[6];
  for (int l = 0; l < 6; ++l)
    for (int r = 0; r < 2; ++r)
      wt_pre[l][r] = (unsigned short*)alloc((size_t)P[l] * a32(27 * P[l]) * 2);
  for (int l = 0; l < 6; ++l) {
    wt_post[l][0] = (unsigned short*)alloc((size_t)P[l] * a32(27 * 2 * P[l]) * 2);
    wt_post[l][1] = (unsigned short*)alloc((size_t)P[l] * a32(27 * P[l]) * 2);
  }
  for (int r = 0; r < 2; ++r)
    wt_inner[r] = (unsigned short*)alloc((size_t)P[6] * a32(27 * P[6]) * 2);
  for (int l = 0; l < 6; ++l) {
    wt_down[l] = (unsigned short*)alloc((size_t)27 * P[l + 1] * a32(P[l]) * 2);
    wt_up[l]   = (unsigned short*)alloc((size_t)27 * P[l] * a32(P[l + 1]) * 2);
  }
  size_t maxel = 0;
  for (int l = 0; l < 6; ++l) { size_t e = (size_t)N[l] * 2 * P[l]; if (e > maxel) maxel = e; }
  { size_t e = (size_t)N[6] * P[6]; if (e > maxel) maxel = e; }
  maxel = (maxel + 7) & ~(size_t)7;
  unsigned short* bufA = (unsigned short*)alloc(maxel * 2);   // bf16 activations
  unsigned short* bufB = (unsigned short*)alloc(maxel * 2);
  float*          accum = (float*)alloc(maxel * 4);           // fp32 scatter accum
  unsigned short* featbf = (unsigned short*)alloc((size_t)N0 * 3 * 2);
  unsigned short* skip[6];
  for (int l = 0; l < 6; ++l) skip[l] = (unsigned short*)alloc((size_t)N[l] * P[l] * 2);

  // Weight repack fp32 -> bf16
  auto repack_full = [&](const float* w, unsigned short* wt, int Cin, int Cout) {
    int Kpad = a32(27 * Cin);
    long long tot = (long long)Cout * Kpad;
    wt_full_kernel<<<dim3((unsigned)((tot + 255) / 256)), dim3(256), 0, stream>>>(
        w, wt, Cin, Cout, Kpad);
  };
  auto repack_perk = [&](const float* w, unsigned short* wt, int Cin, int Cout) {
    int Cpad = a32(Cin);
    long long tot = 27LL * Cout * Cpad;
    wt_perk_kernel<<<dim3((unsigned)((tot + 255) / 256)), dim3(256), 0, stream>>>(
        w, wt, Cin, Cout, Cpad);
  };
  repack_full(Fp(IW), wt_init, 3, 16);
  for (int l = 0; l < 6; ++l)
    for (int r = 0; r < 2; ++r) {
      repack_full(Fp(PRE + (l * 2 + r) * 5), wt_pre[l][r], P[l], P[l]);
      repack_full(Fp(PST + (l * 2 + r) * 5), wt_post[l][r], (r == 0 ? 2 : 1) * P[l], P[l]);
    }
  for (int r = 0; r < 2; ++r) repack_full(Fp(INR + r * 5), wt_inner[r], P[6], P[6]);
  for (int l = 0; l < 6; ++l) {
    repack_perk(Fp(DWN + l * 5), wt_down[l], P[l], P[l + 1]);
    repack_perk(Fp(UPW + l * 5), wt_up[l], P[l + 1], P[l]);
  }

  // Subm conv dispatch (templated on CIN); single LDS buffer when 1 chunk
  auto subm = [&](const unsigned short* x, int nbr_i, const unsigned short* wt, int bn,
                  unsigned short* out, int n, int Cin, int Cout) {
    int Kpad = a32(27 * Cin);
    int nchunks = (Kpad + KCH - 1) / KCH;
    size_t shm = (size_t)(nchunks > 1 ? 2 : 1) * 32 * KCH * 2 + 27 * 32 * 4;
    dim3 grd((unsigned)((n + 31) / 32)), blk(32 * (Cout / 16));
    const int* nb = Ip(nbr_i);
    const float *g = Fp(bn), *b = Fp(bn + 1), *m = Fp(bn + 2), *v = Fp(bn + 3);
#define SUBM_CASE(C) case C: \
    subm_wmma_kernel<C><<<grd, blk, shm, stream>>>(x, nb, wt, g, b, m, v, out, n, Cout, Kpad); break;
    switch (Cin) {
      SUBM_CASE(3)  SUBM_CASE(16)  SUBM_CASE(32)  SUBM_CASE(48)
      SUBM_CASE(64) SUBM_CASE(80)  SUBM_CASE(96)  SUBM_CASE(112)
      SUBM_CASE(128) SUBM_CASE(160) SUBM_CASE(192) SUBM_CASE(224)
      default: break;
    }
#undef SUBM_CASE
  };
  auto zero = [&](float* p, size_t n) {
    size_t n4 = (n + 3) / 4;
    zero_kernel<<<dim3((unsigned)((n4 + 255) / 256)), dim3(256), 0, stream>>>(
        (float4*)p, (long long)n4);
  };
  auto sconv = [&](const unsigned short* x, int g_i, int s_i, const unsigned short* wt2,
                   float* out, int mx, int Cin, int Cout) {
    int tiles = (mx + 15) / 16;
    dim3 grd((unsigned)(27 * tiles)), blk(32 * (Cout / 16));
    const int* gi = Ip(g_i); const int* si = Ip(s_i);
#define SC_CASE(C) case C: \
    scatter_wmma_kernel<C><<<grd, blk, 0, stream>>>(x, gi, si, wt2, out, mx, Cout, tiles); break;
    switch (Cin) {
      SC_CASE(16) SC_CASE(32) SC_CASE(48) SC_CASE(64)
      SC_CASE(80) SC_CASE(96) SC_CASE(112)
      default: break;
    }
#undef SC_CASE
  };
  auto bnstore = [&](const float* acc, int bn, unsigned short* dst,
                     int n, int C, int ld, int coloff) {
    long long tot = (long long)n * C;
    bnrelu_store_kernel<<<dim3((unsigned)((tot + 255) / 256)), dim3(256), 0, stream>>>(
        acc, Fp(bn), Fp(bn + 1), Fp(bn + 2), Fp(bn + 3), dst, n, C, ld, coloff);
  };
  auto catcopy = [&](const unsigned short* src, unsigned short* dst, int n, int C, int ld) {
    long long nvec = (long long)n * (C / 8);
    copy_cols_kernel<<<dim3((unsigned)((nvec + 255) / 256)), dim3(256), 0, stream>>>(
        (const uint4*)src, (uint4*)dst, nvec, C / 8, ld / 8);
  };

  // ---------------- forward ----------------
  {
    long long tot = (long long)N0 * 3;
    f2bf_kernel<<<dim3((unsigned)((tot + 255) / 256)), dim3(256), 0, stream>>>(
        Fp(FEAT), featbf, tot);
  }
  unsigned short* cur = bufA;
  unsigned short* oth = bufB;
  subm(featbf, NBR + 0, wt_init, IW + 1, cur, N0, 3, 16);

  for (int l = 0; l < 6; ++l) {                       // encoder
    subm(cur, NBR + l, wt_pre[l][0], PRE + (l * 2 + 0) * 5 + 1, oth, N[l], P[l], P[l]);
    subm(oth, NBR + l, wt_pre[l][1], PRE + (l * 2 + 1) * 5 + 1, skip[l], N[l], P[l], P[l]);
    zero(accum, (size_t)N[l + 1] * P[l + 1]);
    sconv(skip[l], PIO + 2 * l, PIO + 2 * l + 1, wt_down[l], accum, Mx[l], P[l], P[l + 1]);
    bnstore(accum, DWN + l * 5 + 1, cur, N[l + 1], P[l + 1], P[l + 1], 0);
  }

  subm(cur, NBR + 6, wt_inner[0], INR + 1,     oth, N[6], P[6], P[6]);   // bottleneck
  subm(oth, NBR + 6, wt_inner[1], INR + 5 + 1, cur, N[6], P[6], P[6]);

  for (int l = 5; l >= 0; --l) {                      // decoder
    unsigned short* cat = oth;
    catcopy(skip[l], cat, N[l], P[l], 2 * P[l]);      // concat left half
    zero(accum, (size_t)N[l] * P[l]);
    sconv(cur, PIO + 2 * l + 1, PIO + 2 * l, wt_up[l], accum, Mx[l], P[l + 1], P[l]);
    bnstore(accum, UPW + l * 5 + 1, cat, N[l], P[l], 2 * P[l], P[l]);   // concat right half
    subm(cat, NBR + l, wt_post[l][0], PST + (l * 2 + 0) * 5 + 1, cur, N[l], 2 * P[l], P[l]);
    subm(cur, NBR + l, wt_post[l][1], PST + (l * 2 + 1) * 5 + 1, cat, N[l], P[l], P[l]);
    cur = cat;
    oth = (cur == bufA) ? bufB : bufA;
  }

  {
    long long tot = (long long)N0 * 20;
    linear_kernel<<<dim3((unsigned)((tot + 255) / 256)), dim3(256), 0, stream>>>(
        cur, Fp(LW), Fp(LB), (float*)d_out, N0);
  }
}
